// mLSTMMoELayer_71536975282982
// MI455X (gfx1250) — matrix operand order repaired and checked by reading, not compiled, not run on hardware
//
#include <hip/hip_runtime.h>
#include <hip/hip_bf16.h>

typedef __bf16 bf16_t;
typedef __attribute__((ext_vector_type(16))) __bf16 v16bf;
typedef __attribute__((ext_vector_type(8)))  __bf16 v8bf;
typedef __attribute__((ext_vector_type(8)))  float  v8f;

#define D_MODEL  1024
#define I_DIM    2048
#define NHEADS   4
#define DHEAD    512
#define E_NUM    8
#define T_TOT    8192
#define CHUNK_T  2048
#define LN_EPS   1e-5f
#define CELL_EPS 1e-6f

__device__ __forceinline__ float sigmoidf_(float x) { return 1.0f / (1.0f + __expf(-x)); }

// wave32 butterfly sum
__device__ __forceinline__ float wave_sum(float v) {
#pragma unroll
    for (int m = 16; m >= 1; m >>= 1) v += __shfl_xor(v, m, 32);
    return v;
}

// CDNA5 async global->LDS copy (ASYNCcnt-tracked). lds_off = byte offset in LDS.
__device__ __forceinline__ void async_copy_b128(unsigned int lds_off, const bf16_t* gaddr) {
    asm volatile("global_load_async_to_lds_b128 %0, %1, off"
                 :: "v"(lds_off), "v"(gaddr) : "memory");
}
__device__ __forceinline__ void wait_async_le2() {
    asm volatile("s_wait_asynccnt 2" ::: "memory");
}

__device__ __forceinline__ v16bf load_afrag_g(const bf16_t* p) {   // halves {0..15}
    union { v16bf v; v8bf h[2]; } u;
    u.h[0] = *(const v8bf*)(p);
    u.h[1] = *(const v8bf*)(p + 8);
    return u.v;
}
__device__ __forceinline__ v16bf load_bfrag_lds(const bf16_t* p) { // 16 contiguous halves
    union { v16bf v; v8bf h[2]; } u;
    u.h[0] = *(const v8bf*)(p);
    u.h[1] = *(const v8bf*)(p + 8);
    return u.v;
}

// ---------------------------------------------------------------------------
// prep: per-token LN -> bf16 xhat, router logits/softmax/top2, out = residual.
// One block (256 thr = 8 waves) per token; shuffle-based reductions.
// ---------------------------------------------------------------------------
__global__ void prep_kernel(const float* __restrict__ x, const float* __restrict__ gate_w,
                            bf16_t* __restrict__ xhat, float* __restrict__ we,
                            float* __restrict__ out, float* __restrict__ logits_out) {
    __shared__ float wp[8][2];
    __shared__ float wl[8][E_NUM];
    const int t = blockIdx.x, tid = threadIdx.x;
    const int lane = tid & 31, wv = tid >> 5;
    const float* xr = x + (size_t)t * D_MODEL;

    float xv[4];
#pragma unroll
    for (int i = 0; i < 4; ++i) xv[i] = xr[tid * 4 + i];

    float ps = wave_sum(xv[0] + xv[1] + xv[2] + xv[3]);
    if (lane == 0) wp[wv][0] = ps;
    __syncthreads();
    float mu = 0.f;
#pragma unroll
    for (int w = 0; w < 8; ++w) mu += wp[w][0];
    mu *= (1.0f / D_MODEL);

    float q = 0.f;
#pragma unroll
    for (int i = 0; i < 4; ++i) { float d = xv[i] - mu; q += d * d; }
    q = wave_sum(q);
    if (lane == 0) wp[wv][1] = q;

    float lp[E_NUM];
#pragma unroll
    for (int h = 0; h < E_NUM; ++h) {
        float a = 0.f;
#pragma unroll
        for (int i = 0; i < 4; ++i) a += xv[i] * gate_w[(size_t)(tid * 4 + i) * E_NUM + h];
        lp[h] = wave_sum(a);
    }
    if (lane == 0) {
#pragma unroll
        for (int h = 0; h < E_NUM; ++h) wl[wv][h] = lp[h];
    }
    __syncthreads();

    float var = 0.f;
#pragma unroll
    for (int w = 0; w < 8; ++w) var += wp[w][1];
    const float rs = rsqrtf(var * (1.0f / D_MODEL) + LN_EPS);

#pragma unroll
    for (int i = 0; i < 4; ++i) {
        int c = tid * 4 + i;
        xhat[(size_t)t * D_MODEL + c] = (bf16_t)((xv[i] - mu) * rs);
        out[(size_t)t * D_MODEL + c]  = xv[i];     // residual (sum of top-2 weights == 1)
    }

    if (tid == 0) {
        float l[E_NUM], ex[E_NUM], mx, se = 0.f;
        for (int h = 0; h < E_NUM; ++h) {
            float a = 0.f;
            for (int w = 0; w < 8; ++w) a += wl[w][h];
            l[h] = a;
        }
        mx = l[0];
        for (int h = 1; h < E_NUM; ++h) if (l[h] > mx) mx = l[h];
        for (int h = 0; h < E_NUM; ++h) { ex[h] = __expf(l[h] - mx); se += ex[h]; }
        float inv = 1.0f / se;
        float rw[E_NUM];
        for (int h = 0; h < E_NUM; ++h) rw[h] = ex[h] * inv;
        int i1 = 0; for (int h = 1; h < E_NUM; ++h) if (rw[h] > rw[i1]) i1 = h;
        int i2 = (i1 == 0) ? 1 : 0;
        for (int h = 0; h < E_NUM; ++h) if (h != i1 && rw[h] > rw[i2]) i2 = h;
        float wsum = rw[i1] + rw[i2];
        for (int h = 0; h < E_NUM; ++h) we[(size_t)t * E_NUM + h] = 0.f;
        we[(size_t)t * E_NUM + i1] = rw[i1] / wsum;
        we[(size_t)t * E_NUM + i2] = rw[i2] / wsum;
        for (int h = 0; h < E_NUM; ++h) logits_out[(size_t)t * E_NUM + h] = l[h];
    }
}

// ---------------------------------------------------------------------------
// Weight conversion: fold LN weight into w_up rows, transpose to K-contiguous
// bf16 layout for WMMA B-fragment loads.
// ---------------------------------------------------------------------------
__global__ void convert_wup(const float* __restrict__ wup_e, const float* __restrict__ lnw_e,
                            bf16_t* __restrict__ wupT) {
    int idx = blockIdx.x * 256 + threadIdx.x;       // idx = n*1024 + k
    int n = idx >> 10, k = idx & 1023;
    wupT[idx] = (bf16_t)(wup_e[(size_t)k * (2 * I_DIM) + n] * lnw_e[k]);
}

__global__ void convert_wdown(const float* __restrict__ wd_e, bf16_t* __restrict__ wdT) {
    int idx = blockIdx.x * 256 + threadIdx.x;       // idx = d*2048 + i
    int d = idx >> 11, i = idx & 2047;
    wdT[idx] = (bf16_t)(wd_e[(size_t)i * D_MODEL + d]);
}

// ---------------------------------------------------------------------------
// bf16 WMMA GEMM: C[M x N] (+)= A[M x K] * B^T[N x K]   (both K-contiguous)
// Block = 8 waves, tile 128(M) x 128(N); each wave computes a 32 x 64 patch
// (wave m-index = wave>>1, n-index = wave&1), so every LDS B fragment feeds
// TWO WMMAs (1 ds_load_b128 per WMMA). 4-deep async LDS pipeline for B K-tiles
// (issue i+3 while computing i), ping-ponged A fragment registers.
// MODE 0: C = acc.   MODE 1: C += we[m] * acc (fused MoE epilogue).
// ---------------------------------------------------------------------------
template <int MODE>
__global__ void wmma_gemm_bf16(const bf16_t* __restrict__ A, const bf16_t* __restrict__ B,
                               float* __restrict__ C, const float* __restrict__ wgt,
                               int lda, int ldb, int ldc, int K) {
    __shared__ bf16_t sB[4][128 * 32];              // [buf][n_local * 32 + khalf]

    const int tid  = threadIdx.x;
    const int lane = tid & 31;
    const int wave = tid >> 5;
    const int wm   = wave >> 1;                     // 0..3 -> 32-row band
    const int wn   = wave & 1;                      // 0..1 -> 64-col band
    const int m0 = blockIdx.y * 128 + wm * 32;
    const int n0w = blockIdx.x * 128 + wn * 64;
    const int l15 = lane & 15;
    const int hi  = lane >> 4;
    const int aoff = hi << 4;
    const int ntiles = K >> 5;

    // staging: thread copies 16 halves (32B) of the 128x32 B tile
    const int sn  = tid >> 1;
    const int sko = (tid & 1) << 4;
    const bf16_t* gB = B + (size_t)((blockIdx.x * 128) + sn) * ldb + sko;

    v8f acc0[4] = {};                               // rows m0 .. m0+15
    v8f acc1[4] = {};                               // rows m0+16 .. m0+31
    const bf16_t* arow0 = A + (size_t)(m0 + l15) * lda + aoff;
    const bf16_t* arow1 = arow0 + (size_t)16 * lda;
    const bf16_t* lbase = &sB[0][(size_t)(wn * 64 + l15) * 32 + hi * 16];

    // prologue: issue tiles 0..2, preload A frags for tile 0
#pragma unroll
    for (int tI = 0; tI < 3; ++tI)
        if (tI < ntiles)
            async_copy_b128((unsigned int)(size_t)(&sB[tI][tid * 16]), gB + ((size_t)tI << 5));
    v16bf uaA[2], uaB[2];
    uaA[0] = load_afrag_g(arow0);
    uaA[1] = load_afrag_g(arow1);
    uaB[0] = uaA[0]; uaB[1] = uaA[1];
    wait_async_le2();           // tile 0 resident (in-order completion)
    __syncthreads();

    auto step = [&](int it, v16bf* uaCur, v16bf* uaNxt) {
        const int k0 = it << 5;
        if (it + 3 < ntiles)
            async_copy_b128((unsigned int)(size_t)(&sB[(it + 3) & 3][tid * 16]),
                            gB + ((size_t)(it + 3) << 5));
        if (it + 1 < ntiles) {
            uaNxt[0] = load_afrag_g(arow0 + k0 + 32);
            uaNxt[1] = load_afrag_g(arow1 + k0 + 32);
        }
        __builtin_prefetch(arow0 + k0 + 96, 0, 3);

        const bf16_t* lb = lbase + (size_t)(it & 3) * (128 * 32);
        v16bf ub0 = load_bfrag_lds(lb);
        v16bf ub1;
#pragma unroll
        for (int j = 0; j < 4; ++j) {
            v16bf& cur = (j & 1) ? ub1 : ub0;
            v16bf& nxt = (j & 1) ? ub0 : ub1;
            if (j < 3) nxt = load_bfrag_lds(lb + (size_t)(j + 1) * 512);
            acc0[j] = __builtin_amdgcn_wmma_f32_16x16x32_bf16(
                false, uaCur[0], false, cur, (short)0, acc0[j], false, false);
            acc1[j] = __builtin_amdgcn_wmma_f32_16x16x32_bf16(
                false, uaCur[1], false, cur, (short)0, acc1[j], false, false);
        }
        wait_async_le2();       // tile it+1 resident; it+2/it+3 may stay in flight
        __syncthreads();        // all waves done reading buf(it) / filling buf(it+3)
    };

    for (int it = 0; it < ntiles; it += 2) {        // ntiles is even (K mult. of 64)
        step(it,     uaA, uaB);
        step(it + 1, uaB, uaA);
    }

#pragma unroll
    for (int j = 0; j < 4; ++j) {
        const int n = n0w + j * 16 + l15;
#pragma unroll
        for (int r = 0; r < 8; ++r) {
            const int mA = m0 + r + (hi << 3);          // C/D: VGPR r -> row r / r+8
            const int mB = mA + 16;
            const float vA = acc0[j][r];
            const float vB = acc1[j][r];
            if (MODE == 0) {
                C[(size_t)mA * ldc + n] = vA;
                C[(size_t)mB * ldc + n] = vB;
            } else {
                const float wA = wgt[(size_t)mA * E_NUM];
                const float wB = wgt[(size_t)mB * E_NUM];
                C[(size_t)mA * ldc + n] += wA * vA;
                C[(size_t)mB * ldc + n] += wB * vB;
            }
        }
    }
}

// ---------------------------------------------------------------------------
// mid: conv-scale + silu, 4x4 headwise q/k/v, input-gate, mLSTM cell scale,
//      per-head LN, skip & z-gate -> hs (bf16). One block (256 thr) per token.
//      Wave-shuffle reductions; head h owns waves 2h, 2h+1.
// ---------------------------------------------------------------------------
__global__ void mid_kernel(const float* __restrict__ up,   // [CHUNK_T][2*I]
                           bf16_t* __restrict__ hsb,       // [CHUNK_T][I]
                           const float* __restrict__ conv_w, const float* __restrict__ conv_b,
                           const float* __restrict__ q_w, const float* __restrict__ k_w,
                           const float* __restrict__ v_w, const float* __restrict__ ig_w,
                           const float* __restrict__ ig_b, const float* __restrict__ on_w,
                           const float* __restrict__ skip_w, int e) {
    __shared__ float wpart[8][5];       // [wave][0..3]=ig partials, [4]=qk partial
    __shared__ float wms[8][2];         // [wave][0]=sum, [1]=sumsq
    __shared__ float s_scale[NHEADS];
    const int row = blockIdx.x, tid = threadIdx.x;
    const int lane = tid & 31, wv = tid >> 5;
    const int c0 = tid * 8;

    const float* upr = up + (size_t)row * (2 * I_DIM);
    const float* cw  = conv_w + (size_t)e * I_DIM * 4;
    const float* cb  = conv_b + (size_t)e * I_DIM;
    const float* qw  = q_w + (size_t)e * (I_DIM / 4) * 16;
    const float* kw  = k_w + (size_t)e * (I_DIM / 4) * 16;
    const float* vw  = v_w + (size_t)e * (I_DIM / 4) * 16;
    const float* igw = ig_w + (size_t)e * 3 * I_DIM * NHEADS;
    const float* igb = ig_b + (size_t)e * NHEADS;
    const float* onw = on_w + (size_t)e * I_DIM;
    const float* skw = skip_w + (size_t)e * I_DIM;

    float xm[8], zz[8], ca[8], qv[8], kv[8], vv[8];
#pragma unroll
    for (int i = 0; i < 8; ++i) { xm[i] = upr[c0 + i]; zz[i] = upr[I_DIM + c0 + i]; }
#pragma unroll
    for (int i = 0; i < 8; ++i) {
        int c = c0 + i;
        float cv = xm[i] * cw[c * 4 + 3] + cb[c];
        ca[i] = cv * sigmoidf_(cv);
    }
#pragma unroll
    for (int g = 0; g < 2; ++g) {
        int n = (c0 >> 2) + g;
        const float* qwn = qw + n * 16;
        const float* kwn = kw + n * 16;
        const float* vwn = vw + n * 16;
#pragma unroll
        for (int o = 0; o < 4; ++o) {
            float sq = 0.f, sk = 0.f, sv = 0.f;
#pragma unroll
            for (int b = 0; b < 4; ++b) {
                sq += qwn[o * 4 + b] * ca[g * 4 + b];
                sk += kwn[o * 4 + b] * ca[g * 4 + b];
                sv += vwn[o * 4 + b] * xm[g * 4 + b];
            }
            qv[g * 4 + o] = sq; kv[g * 4 + o] = sk; vv[g * 4 + o] = sv;
        }
    }
    float igp[NHEADS] = {0.f, 0.f, 0.f, 0.f};
#pragma unroll
    for (int i = 0; i < 8; ++i) {
        int c = c0 + i;
#pragma unroll
        for (int h = 0; h < NHEADS; ++h)
            igp[h] += qv[i] * igw[(size_t)c * 4 + h]
                    + kv[i] * igw[(size_t)(I_DIM + c) * 4 + h]
                    + vv[i] * igw[(size_t)(2 * I_DIM + c) * 4 + h];
    }
    float qkp = 0.f;
#pragma unroll
    for (int i = 0; i < 8; ++i) qkp += qv[i] * kv[i];
#pragma unroll
    for (int h = 0; h < NHEADS; ++h) igp[h] = wave_sum(igp[h]);
    qkp = wave_sum(qkp);
    if (lane == 0) {
#pragma unroll
        for (int h = 0; h < NHEADS; ++h) wpart[wv][h] = igp[h];
        wpart[wv][4] = qkp;
    }
    __syncthreads();

    if (tid < NHEADS) {
        float ig = igb[tid];
#pragma unroll
        for (int w = 0; w < 8; ++w) ig += wpart[w][tid];
        float qk  = (wpart[2 * tid][4] + wpart[2 * tid + 1][4]) * 0.044194173824159216f;
        float nrm = fmaxf(fabsf(qk), __expf(-ig));
        s_scale[tid] = qk / (nrm + CELL_EPS);
    }
    __syncthreads();

    const int head = wv >> 1;
    const float sc = s_scale[head];
    float hv[8]; float sum = 0.f, ssq = 0.f;
#pragma unroll
    for (int i = 0; i < 8; ++i) { hv[i] = sc * vv[i]; sum += hv[i]; ssq += hv[i] * hv[i]; }
    sum = wave_sum(sum);
    ssq = wave_sum(ssq);
    if (lane == 0) { wms[wv][0] = sum; wms[wv][1] = ssq; }
    __syncthreads();

    const float hsum = wms[2 * head][0] + wms[2 * head + 1][0];
    const float hssq = wms[2 * head][1] + wms[2 * head + 1][1];
    const float mu = hsum * (1.0f / DHEAD);
    const float rs = rsqrtf(hssq * (1.0f / DHEAD) - mu * mu + LN_EPS);

#pragma unroll
    for (int i = 0; i < 8; ++i) {
        int c = c0 + i;
        float hn  = (hv[i] - mu) * rs * onw[c];
        float sz  = zz[i] * sigmoidf_(zz[i]);
        float o   = (hn + skw[c] * ca[i]) * sz;
        hsb[(size_t)row * I_DIM + c] = (bf16_t)o;
    }
}

// ---------------------------------------------------------------------------
extern "C" void kernel_launch(void* const* d_in, const int* in_sizes, int n_in,
                              void* d_out, int out_size, void* d_ws, size_t ws_size,
                              hipStream_t stream) {
    (void)in_sizes; (void)n_in; (void)out_size; (void)ws_size;
    const float* x      = (const float*)d_in[0];
    const float* gate_w = (const float*)d_in[1];
    const float* ln_w   = (const float*)d_in[2];
    const float* w_up   = (const float*)d_in[3];
    const float* conv_w = (const float*)d_in[4];
    const float* conv_b = (const float*)d_in[5];
    const float* q_w    = (const float*)d_in[6];
    const float* k_w    = (const float*)d_in[7];
    const float* v_w    = (const float*)d_in[8];
    const float* ig_w   = (const float*)d_in[9];
    const float* ig_b   = (const float*)d_in[10];
    // d_in[11], d_in[12] = fg_w, fg_b: unused by the reference math
    const float* on_w   = (const float*)d_in[13];
    const float* skip_w = (const float*)d_in[14];
    const float* w_down = (const float*)d_in[15];

    float* out    = (float*)d_out;                     // [T_TOT][D_MODEL]
    float* logits = out + (size_t)T_TOT * D_MODEL;     // [T_TOT][E_NUM]

    char* ws = (char*)d_ws;
    bf16_t* xhat = (bf16_t*)ws;  ws += (size_t)T_TOT * D_MODEL * sizeof(bf16_t);
    bf16_t* wupT = (bf16_t*)ws;  ws += (size_t)(2 * I_DIM) * D_MODEL * sizeof(bf16_t);
    bf16_t* wdT  = (bf16_t*)ws;  ws += (size_t)D_MODEL * I_DIM * sizeof(bf16_t);
    float*  we   = (float*)ws;   ws += (size_t)T_TOT * E_NUM * sizeof(float);
    float*  up   = (float*)ws;   ws += (size_t)CHUNK_T * (2 * I_DIM) * sizeof(float);
    bf16_t* hsb  = (bf16_t*)ws;  ws += (size_t)CHUNK_T * I_DIM * sizeof(bf16_t);

    prep_kernel<<<T_TOT, 256, 0, stream>>>(x, gate_w, xhat, we, out, logits);

    for (int e = 0; e < E_NUM; ++e) {
        convert_wup<<<(2 * I_DIM * D_MODEL) / 256, 256, 0, stream>>>(
            w_up + (size_t)e * D_MODEL * (2 * I_DIM), ln_w + (size_t)e * D_MODEL, wupT);
        convert_wdown<<<(D_MODEL * I_DIM) / 256, 256, 0, stream>>>(
            w_down + (size_t)e * I_DIM * D_MODEL, wdT);

        for (int ch = 0; ch < T_TOT / CHUNK_T; ++ch) {
            const bf16_t* Ach = xhat + (size_t)ch * CHUNK_T * D_MODEL;
            // up-projection: [CHUNK_T x 1024] @ [1024 x 4096]
            wmma_gemm_bf16<0><<<dim3((2 * I_DIM) / 128, CHUNK_T / 128), 256, 0, stream>>>(
                Ach, wupT, up, nullptr, D_MODEL, D_MODEL, 2 * I_DIM, D_MODEL);
            // elementwise / mLSTM cell
            mid_kernel<<<CHUNK_T, 256, 0, stream>>>(up, hsb, conv_w, conv_b, q_w, k_w, v_w,
                                                    ig_w, ig_b, on_w, skip_w, e);
            // down-projection with fused MoE-weighted accumulate into out
            wmma_gemm_bf16<1><<<dim3(D_MODEL / 128, CHUNK_T / 128), 256, 0, stream>>>(
                hsb, wdT, out + (size_t)ch * CHUNK_T * D_MODEL,
                we + (size_t)ch * CHUNK_T * E_NUM + e, I_DIM, I_DIM, D_MODEL, I_DIM);
        }
    }
}
